// HashingSymbol_42245298323618
// MI455X (gfx1250) — matrix operand order repaired and checked by reading, not compiled, not run on hardware
//
#include <hip/hip_runtime.h>
#include <hip/hip_bf16.h>

typedef __attribute__((ext_vector_type(16))) __bf16 bf16x16;
typedef __attribute__((ext_vector_type(8)))  float  f32x8;
typedef unsigned short u16;
typedef unsigned int   u32;
typedef unsigned long long u64;
typedef __attribute__((ext_vector_type(4))) u32 u32x4;
typedef __attribute__((ext_vector_type(8))) int i32x8;
typedef __attribute__((ext_vector_type(4))) int i32x4;

#define B_   4
#define S_   512
#define D_   512
#define NS_  8
#define NQ_  16
#define NK_  2048
#define KD_  64
#define VD_  128
#define MEM_ 512
#define BS_  (B_*S_)   // 2048 tokens

#if defined(__has_builtin)
#if __has_builtin(__builtin_amdgcn_tensor_load_to_lds) && __has_builtin(__builtin_amdgcn_s_wait_tensorcnt)
#define HAVE_TDM 1
#endif
#endif
#ifndef HAVE_TDM
#define HAVE_TDM 0
#endif

union ABPack { bf16x16 v; u32 d[8]; u16 s[16]; };

__device__ __forceinline__ u16 f2bf(float f) {
  u32 u = __builtin_bit_cast(u32, f);
  u32 r = u + 0x7FFFu + ((u >> 16) & 1u);   // round-to-nearest-even
  return (u16)(r >> 16);
}

// A fragment: 16(M) x 32(K) bf16. ISA layout: lane lh (0..15) holds row M=lh,
// kg=lane>>4 selects K half; VGPR i packs K = (i&3)*2 + kg*8 + (i>>2)*16 + {0,1}.
__device__ __forceinline__ bf16x16 load_a_tile(const u16* p, int row_stride) {
  int lane = threadIdx.x & 31, lh = lane & 15, kg = lane >> 4;
  const u16* q = p + lh * row_stride + kg * 8;
  ABPack r;
#pragma unroll
  for (int i = 0; i < 8; ++i) {
    int off = (i & 3) * 2 + (i >> 2) * 16;
    r.d[i] = *(const u32*)(q + off);
  }
  return r.v;
}

// B fragment: 32(K) x 16(N) bf16, source stored N-major (B^T, K contiguous):
// lane lh holds col N=lh; VGPR i packs K = (i&3)*2 + kg*8 + (i>>2)*16 + {0,1}.
__device__ __forceinline__ bf16x16 load_b_tile(const u16* p, int col_stride) {
  int lane = threadIdx.x & 31, lh = lane & 15, kg = lane >> 4;
  const u16* q = p + lh * col_stride + kg * 8;
  ABPack r;
#pragma unroll
  for (int i = 0; i < 8; ++i) r.d[i] = *(const u32*)(q + (i & 3) * 2 + (i >> 2) * 16);
  return r.v;
}

__device__ __forceinline__ f32x8 wmma_bf16(bf16x16 a, bf16x16 b, f32x8 c) {
  return __builtin_amdgcn_wmma_f32_16x16x32_bf16(false, a, false, b, (short)0, c, false, false);
}

__device__ __forceinline__ u32 lds_off(const void* p) {
  // generic pointers to LDS carry the wave-relative LDS byte offset in the low 32 bits
  return (u32)(uintptr_t)p;
}

#if HAVE_TDM
// Build a 2D tensor-DMA descriptor (D#) per CDNA5 ISA §8.3/8.4 and issue
// TENSOR_LOAD_TO_LDS. Copies tile1 rows x tile0 cols of 2-byte elements,
// row pitch = stride0 elements, contiguous into LDS at lds.
__device__ __forceinline__ void tdm_load_2d(u32 lds, const void* gsrc,
                                            u32 tdim0, u32 tdim1, u32 stride0,
                                            u32 tile0, u32 tile1) {
  u64 ga = (u64)(uintptr_t)gsrc;
  u32x4 g0;
  g0[0] = 1u;                                            // count=1 (valid), user D#
  g0[1] = lds;                                           // lds_addr (bytes)
  g0[2] = (u32)ga;                                       // global_addr[31:0]
  g0[3] = (u32)((ga >> 32) & 0x1FFFFFFu) | 0x80000000u;  // global_addr[56:32] | type=2<<30
  i32x8 g1;
  g1[0] = (int)(1u << 16);                               // workgroup_mask=0, data_size=1 (2B)
  g1[1] = (int)((tdim0 & 0xFFFFu) << 16);                // tensor_dim0[15:0] @ bits[63:48]
  g1[2] = (int)((tdim0 >> 16) | ((tdim1 & 0xFFFFu) << 16)); // dim0 hi | dim1 lo
  g1[3] = (int)((tdim1 >> 16) | (tile0 << 16));          // dim1 hi | tile_dim0
  g1[4] = (int)tile1;                                    // tile_dim1 | tile_dim2=0
  g1[5] = (int)stride0;                                  // tensor_dim0_stride[31:0]
  g1[6] = 0;                                             // stride0 hi | stride1 lo (unused, 2D)
  g1[7] = 0;
  i32x4 z4 = {0, 0, 0, 0};
#if __clang_major__ >= 23
  i32x8 z8 = {0, 0, 0, 0, 0, 0, 0, 0};
  __builtin_amdgcn_tensor_load_to_lds(g0, g1, z4, z4, z8, 0);
#else
  __builtin_amdgcn_tensor_load_to_lds(g0, g1, z4, z4, 0);
#endif
}
#else
// Fallback: cooperative global->LDS stage of a 64x32 bf16 tile (256 threads)
__device__ __forceinline__ void stage_tile_64x32(u16* dst, const u16* src, int row_stride) {
  for (int i = threadIdx.x; i < 1024; i += 256) {
    int r = i >> 4, c = (i & 15) * 2;
    *(u32*)(dst + r * 32 + c) = *(const u32*)(src + (size_t)r * row_stride + c);
  }
}
#endif

// ---------------- prep kernels: fp32 -> bf16 (+transpose / key-normalize) ----
__global__ void cvt_bf16_kernel(const float* __restrict__ src, u16* __restrict__ dst, int n) {
  int i = blockIdx.x * blockDim.x + threadIdx.x;
  if (i < n) dst[i] = f2bf(src[i]);
}

// src is K x N row-major; dst is N x K (K contiguous) bf16
__global__ void transpose_bf16_kernel(const float* __restrict__ src, u16* __restrict__ dst,
                                      int K, int N) {
  int i = blockIdx.x * blockDim.x + threadIdx.x;
  if (i < K * N) {
    int n = i / K, k = i % K;
    dst[i] = f2bf(src[(size_t)k * N + n]);
  }
}

__global__ void knorm_kernel(const float* __restrict__ keys, u16* __restrict__ khat) {
  int i = blockIdx.x * blockDim.x + threadIdx.x;
  if (i < NS_ * NK_) {
    const float* kp = keys + (size_t)i * KD_;
    float s = 0.f;
#pragma unroll 8
    for (int e = 0; e < KD_; ++e) { float v = kp[e]; s += v * v; }
    float rs = rsqrtf(s);
    u16* op = khat + (size_t)i * KD_;
#pragma unroll 8
    for (int e = 0; e < KD_; ++e) op[e] = f2bf(kp[e] * rs);
  }
}

// ---------------- q = LN(x @ Wq + bq) --------------------------------------
// 64x64 tile per block; 8 waves, each owns a 16-row strip and 2 16-col subtiles.
// A/B 64x32 bf16 tiles are DMA'd into LDS by the Tensor Data Mover (wave 0
// issues the D#, waits on TENSORcnt, workgroup barrier publishes), double-
// buffered so the DMA of step k+1 overlaps the WMMAs of step k.
// N-tile of 64 == exactly one KD layernorm group, so LN is done in-tile.
__global__ void qproj_ln_kernel(const u16* __restrict__ xb, const u16* __restrict__ WqT,
                                const float* __restrict__ bq, const float* __restrict__ ln_g,
                                const float* __restrict__ ln_b, u16* __restrict__ qb) {
  __shared__ u16 bufA[2][64 * 32];
  __shared__ u16 bufB[2][64 * 32];
  __shared__ float Ct[64][65];
  int m0 = blockIdx.x * 64, n0 = blockIdx.y * 64;
  int wave = threadIdx.x >> 5;
  int lane = threadIdx.x & 31, lh = lane & 15, kg = lane >> 4;
  int sr = wave & 3;
  int sc0 = (wave >> 2) * 2;
  const u16* Abase = xb + (size_t)m0 * D_;
  const u16* Bbase = WqT + (size_t)n0 * D_;

#if HAVE_TDM
  if (wave == 0) {
    tdm_load_2d(lds_off(bufA[0]), Abase, D_, BS_, D_, 32, 64);
    tdm_load_2d(lds_off(bufB[0]), Bbase, D_, NQ_ * KD_, D_, 32, 64);
    __builtin_amdgcn_s_wait_tensorcnt(0);
  }
#else
  stage_tile_64x32(bufA[0], Abase, D_);
  stage_tile_64x32(bufB[0], Bbase, D_);
#endif
  __syncthreads();

  f32x8 acc0 = {}, acc1 = {};
  int buf = 0;
  for (int ks = 0; ks < 16; ++ks) {
#if HAVE_TDM
    if (wave == 0 && ks < 15) {     // prefetch next K-slab into the other buffer
      int k1 = (ks + 1) * 32;
      tdm_load_2d(lds_off(bufA[buf ^ 1]), Abase + k1, D_, BS_, D_, 32, 64);
      tdm_load_2d(lds_off(bufB[buf ^ 1]), Bbase + k1, D_, NQ_ * KD_, D_, 32, 64);
    }
#endif
    bf16x16 a  = load_a_tile(&bufA[buf][sr * 16 * 32], 32);
    bf16x16 b0 = load_b_tile(&bufB[buf][sc0 * 16 * 32], 32);
    bf16x16 b1 = load_b_tile(&bufB[buf][(sc0 + 1) * 16 * 32], 32);
    acc0 = wmma_bf16(a, b0, acc0);
    acc1 = wmma_bf16(a, b1, acc1);
#if HAVE_TDM
    if (wave == 0 && ks < 15) __builtin_amdgcn_s_wait_tensorcnt(0);
    __syncthreads();
#else
    __syncthreads();
    if (ks < 15) {
      int k1 = (ks + 1) * 32;
      stage_tile_64x32(bufA[buf ^ 1], Abase + k1, D_);
      stage_tile_64x32(bufB[buf ^ 1], Bbase + k1, D_);
      __syncthreads();
    }
#endif
    buf ^= 1;
  }

  float bias0 = bq[n0 + sc0 * 16 + lh];
  float bias1 = bq[n0 + (sc0 + 1) * 16 + lh];
#pragma unroll
  for (int r = 0; r < 8; ++r) {
    int row = sr * 16 + r + kg * 8;
    Ct[row][sc0 * 16 + lh]       = acc0[r] + bias0;
    Ct[row][(sc0 + 1) * 16 + lh] = acc1[r] + bias1;
  }
  __syncthreads();
  if (threadIdx.x < 64) {
    int row = threadIdx.x;
    float s = 0.f, s2 = 0.f;
#pragma unroll 8
    for (int c = 0; c < 64; ++c) { float v = Ct[row][c]; s += v; s2 += v * v; }
    float mu  = s * (1.f / 64.f);
    float var = s2 * (1.f / 64.f) - mu * mu;
    float rs  = rsqrtf(var + 1e-5f);
    u16* dst = qb + (size_t)(m0 + row) * (NQ_ * KD_) + n0;
#pragma unroll 8
    for (int c = 0; c < 64; ++c)
      dst[c] = f2bf((Ct[row][c] - mu) * rs * ln_g[c] + ln_b[c]);
  }
}

// ---------------- scores -> top8 -> softmax -> sparse A@V -------------------
// Block = 16 tokens x 1 slot. Scores 16x2048 computed by WMMA in 4 passes of
// 512 cols held in LDS; per-thread register top-8 (insertion network) merged
// via LDS; softmax over the 8 survivors (exact: exp(-1e6) == 0 in fp32);
// sparse gather of values (8 of 2048 rows) instead of the dense 16x2048@2048x128.
__global__ void attn_kernel(const u16* __restrict__ qb, const u16* __restrict__ khat,
                            const float* __restrict__ values, float* __restrict__ attn) {
  __shared__ float sc[16][512];
  __shared__ float cv[16][16][8];
  __shared__ int   ci[16][16][8];
  __shared__ float w8[16][8];
  __shared__ int   i8s[16][8];

  int m0 = blockIdx.x * 16;
  int slot = blockIdx.y;
  int ek = slot & (NS_ - 1);                 // keys/values are concat(keys, keys[:8])
  int wave = threadIdx.x >> 5;
  int lane = threadIdx.x & 31, lh = lane & 15, kg = lane >> 4;
  int row = threadIdx.x >> 4;                // scan/gather row 0..15
  int sub = threadIdx.x & 15;

  const u16* Aorig = qb + ((size_t)m0 * NQ_ + slot) * KD_;
  bf16x16 a0 = load_a_tile(Aorig, NQ_ * KD_);
  bf16x16 a1 = load_a_tile(Aorig + 32, NQ_ * KD_);

  float tv[8]; int tix[8];
#pragma unroll
  for (int k = 0; k < 8; ++k) { tv[k] = -1e30f; tix[k] = 0; }

  for (int pass = 0; pass < 4; ++pass) {
    int cb = pass * 512;
#pragma unroll
    for (int cc = 0; cc < 4; ++cc) {
      int jloc = wave * 64 + cc * 16;
      const u16* Bp = khat + ((size_t)ek * NK_ + cb + jloc) * KD_;
      __builtin_prefetch(Bp + 16 * KD_, 0, 0);
      f32x8 acc = {};
      acc = wmma_bf16(a0, load_b_tile(Bp, KD_), acc);
      acc = wmma_bf16(a1, load_b_tile(Bp + 32, KD_), acc);
#pragma unroll
      for (int r = 0; r < 8; ++r)
        sc[r + kg * 8][jloc + lh] = acc[r];
    }
    __syncthreads();
    for (int c = sub * 32; c < sub * 32 + 32; ++c) {
      float v = sc[row][c];
      if (v > tv[7]) {
        tv[7] = v; tix[7] = cb + c;
#pragma unroll
        for (int k = 7; k > 0; --k)
          if (tv[k] > tv[k - 1]) {
            float t = tv[k]; tv[k] = tv[k - 1]; tv[k - 1] = t;
            int ti = tix[k]; tix[k] = tix[k - 1]; tix[k - 1] = ti;
          }
      }
    }
    __syncthreads();
  }
#pragma unroll
  for (int k = 0; k < 8; ++k) { cv[row][sub][k] = tv[k]; ci[row][sub][k] = tix[k]; }
  __syncthreads();

  if (threadIdx.x < 16) {
    int rr = threadIdx.x;
    float mv[8]; int mi[8];
#pragma unroll
    for (int k = 0; k < 8; ++k) { mv[k] = -1e30f; mi[k] = 0; }
    for (int s2 = 0; s2 < 16; ++s2)
#pragma unroll
      for (int k2 = 0; k2 < 8; ++k2) {
        float v = cv[rr][s2][k2]; int ix = ci[rr][s2][k2];
        if (v > mv[7]) {
          mv[7] = v; mi[7] = ix;
#pragma unroll
          for (int k = 7; k > 0; --k)
            if (mv[k] > mv[k - 1]) {
              float t = mv[k]; mv[k] = mv[k - 1]; mv[k - 1] = t;
              int ti = mi[k]; mi[k] = mi[k - 1]; mi[k - 1] = ti;
            }
        }
      }
    float mx = mv[0], sum = 0.f, e[8];
#pragma unroll
    for (int k = 0; k < 8; ++k) { e[k] = __expf(mv[k] - mx); sum += e[k]; }
    float inv = 1.f / sum;
#pragma unroll
    for (int k = 0; k < 8; ++k) { w8[rr][k] = e[k] * inv; i8s[rr][k] = mi[k]; }
  }
  __syncthreads();

  {
    int vb = sub * 8;
    float acc[8];
#pragma unroll
    for (int e2 = 0; e2 < 8; ++e2) acc[e2] = 0.f;
#pragma unroll
    for (int k = 0; k < 8; ++k) {
      float wgt = w8[row][k];
      int j = i8s[row][k];
      const float* vp = values + ((size_t)ek * NK_ + j) * VD_ + vb;
#pragma unroll
      for (int e2 = 0; e2 < 8; ++e2) acc[e2] += wgt * vp[e2];
    }
    float* op = attn + ((size_t)(m0 + row) * NQ_ + slot) * VD_ + vb;
#pragma unroll
    for (int e2 = 0; e2 < 8; ++e2) op[e2] = acc[e2];
  }
}

// ---------------- fused (attn + q@Wres + bres) @ Wmem + bmem ----------------
// One block per token: the 16-row tile is exactly token x all 16 slots, so the
// slot-split writeout (out[:,:,:8] / out[:,:,8:]) falls out of the C layout.
__global__ void resmem_kernel(const u16* __restrict__ qb, const u16* __restrict__ WresT,
                              const float* __restrict__ bres, const float* __restrict__ attn,
                              const u16* __restrict__ WmemT, const float* __restrict__ bmem,
                              float* __restrict__ out0, float* __restrict__ out1) {
  __shared__ u16 t[16][128];
  int token = blockIdx.x;
  int wave = threadIdx.x >> 5;
  int lane = threadIdx.x & 31, lh = lane & 15, kg = lane >> 4;

  { // stage 1: t = bf16(attn + q @ WresT + bres), one 16-col tile per wave
    int n0 = wave * 16;
    f32x8 acc;
#pragma unroll
    for (int r = 0; r < 8; ++r)
      acc[r] = attn[((size_t)token * NQ_ + (r + kg * 8)) * VD_ + n0 + lh];
    const u16* Aorig = qb + (size_t)token * NQ_ * KD_;
    bf16x16 a0 = load_a_tile(Aorig, KD_);
    bf16x16 a1 = load_a_tile(Aorig + 32, KD_);
    const u16* Bp = WresT + (size_t)n0 * KD_;
    acc = wmma_bf16(a0, load_b_tile(Bp, KD_), acc);
    acc = wmma_bf16(a1, load_b_tile(Bp + 32, KD_), acc);
    float bias = bres[n0 + lh];
#pragma unroll
    for (int r = 0; r < 8; ++r)
      t[r + kg * 8][n0 + lh] = f2bf(acc[r] + bias);
  }
  __syncthreads();

  // stage 2: out = t @ WmemT + bmem; A fragments read from LDS
  const u16* Ap = &t[0][0];
  bf16x16 a[4];
#pragma unroll
  for (int kk = 0; kk < 4; ++kk) a[kk] = load_a_tile(Ap + kk * 32, VD_);
#pragma unroll
  for (int cc = 0; cc < 4; ++cc) {
    int n0 = wave * 64 + cc * 16;
    const u16* Bp = WmemT + (size_t)n0 * VD_;
    f32x8 acc = {};
#pragma unroll
    for (int kk = 0; kk < 4; ++kk)
      acc = wmma_bf16(a[kk], load_b_tile(Bp + kk * 32, VD_), acc);
    float bias = bmem[n0 + lh];
#pragma unroll
    for (int r = 0; r < 8; ++r) {
      int s = r + kg * 8;
      float v = acc[r] + bias;
      if (s < NS_)
        out0[((size_t)token * NS_ + s) * MEM_ + n0 + lh] = v;
      else
        out1[((size_t)token * NS_ + (s - NS_)) * MEM_ + n0 + lh] = v;
    }
  }
}

extern "C" void kernel_launch(void* const* d_in, const int* in_sizes, int n_in,
                              void* d_out, int out_size, void* d_ws, size_t ws_size,
                              hipStream_t stream) {
  const float* x      = (const float*)d_in[0];
  const float* Wq     = (const float*)d_in[1];
  const float* bq     = (const float*)d_in[2];
  const float* ln_g   = (const float*)d_in[3];
  const float* ln_b   = (const float*)d_in[4];
  const float* keys   = (const float*)d_in[5];
  const float* values = (const float*)d_in[6];
  const float* Wres   = (const float*)d_in[7];
  const float* bres   = (const float*)d_in[8];
  const float* Wmem   = (const float*)d_in[9];
  const float* bmem   = (const float*)d_in[10];
  float* out0 = (float*)d_out;
  float* out1 = out0 + (size_t)B_ * S_ * NS_ * MEM_;

  char* w = (char*)d_ws;
  u16*   xb    = (u16*)(w);                 // x bf16:      2048*512   (2 MB)
  u16*   WqT   = (u16*)(w + 2097152);       // Wq^T bf16:   1024*512   (1 MB)
  u16*   qb    = (u16*)(w + 3145728);       // q bf16:      2048*1024  (4 MB)
  u16*   khat  = (u16*)(w + 7340032);       // keys/|k|:    8*2048*64  (2 MB)
  u16*   WresT = (u16*)(w + 9437184);       // Wres^T:      128*64
  u16*   WmemT = (u16*)(w + 9453568);       // Wmem^T:      512*128
  float* attn  = (float*)(w + 9584640);     // attn out:    2048*16*128 (16 MB)

  cvt_bf16_kernel<<<(BS_ * D_ + 255) / 256, 256, 0, stream>>>(x, xb, BS_ * D_);
  transpose_bf16_kernel<<<(D_ * NQ_ * KD_ + 255) / 256, 256, 0, stream>>>(Wq, WqT, D_, NQ_ * KD_);
  transpose_bf16_kernel<<<(KD_ * VD_ + 255) / 256, 256, 0, stream>>>(Wres, WresT, KD_, VD_);
  transpose_bf16_kernel<<<(VD_ * MEM_ + 255) / 256, 256, 0, stream>>>(Wmem, WmemT, VD_, MEM_);
  knorm_kernel<<<(NS_ * NK_ + 255) / 256, 256, 0, stream>>>(keys, khat);

  qproj_ln_kernel<<<dim3(BS_ / 64, (NQ_ * KD_) / 64), 256, 0, stream>>>(xb, WqT, bq, ln_g, ln_b, qb);
  attn_kernel<<<dim3(BS_ / 16, NQ_), 256, 0, stream>>>(qb, khat, values, attn);
  resmem_kernel<<<BS_, 256, 0, stream>>>(qb, WresT, bres, attn, WmemT, bmem, out0, out1);
}
